// S7LlmMoe300M_30142080483616
// MI455X (gfx1250) — matrix operand order repaired and checked by reading, not compiled, not run on hardware
//
#include <hip/hip_runtime.h>
#include <math.h>

typedef __attribute__((ext_vector_type(16))) _Float16 v16h;
typedef __attribute__((ext_vector_type(8)))  float    v8f;
typedef __attribute__((ext_vector_type(4)))  float    f4;
typedef __attribute__((ext_vector_type(4)))  _Float16 h4;
typedef __attribute__((ext_vector_type(2)))  _Float16 h2;

#define TBM 64
#define TBN 64
#define TBK 32
#define LPAD 8   // LDS row padding (halves): row = 40 halves = 80 bytes (8B aligned)

__device__ __forceinline__ float gelu_tanh(float x) {
  float x3 = x * x * x;
  return 0.5f * x * (1.0f + tanhf(0.7978845608028654f * (x + 0.044715f * x3)));
}

// ---------------------------------------------------------------------------
// Generic batched WMMA GEMM:  C = act(alpha * A@B (+bias)) (+res)
// A: [M,K] f32 (lda), B: [K,N] f32 (ldb) or transB -> Bmem is [N,K] row-major.
// Batched over grid.z: off = (z/Hdim)*s?b + (z%Hdim)*s?h  (element strides).
// f16 compute via v_wmma_f32_16x16x32_f16, f32 accumulate.
// Fast path (full tiles, K%32==0): float4 global loads, register-prefetch
// double buffering, b64 LDS stores, b32 LDS fragment reads.
// ---------------------------------------------------------------------------
__global__ __launch_bounds__(128) void k_gemm(
    const float* __restrict__ A, const float* __restrict__ Bm, float* __restrict__ C,
    const float* __restrict__ bias, const float* __restrict__ res,
    int M, int N, int K, int lda, int ldb, int ldc,
    long long sAb, long long sAh, long long sBb, long long sBh,
    long long sCb, long long sCh,
    int Hdim, int transB, int actGelu, float alpha)
{
  // K-minor LDS layouts for both operands -> pair (b32) fragment reads
  __shared__ _Float16 As[TBM][TBK + LPAD];   // As[m][k]
  __shared__ _Float16 Bt[TBN][TBK + LPAD];   // Bt[n][k]

  int z  = blockIdx.z;
  int bb = z / Hdim, hh = z % Hdim;
  A  += (long long)bb * sAb + (long long)hh * sAh;
  Bm += (long long)bb * sBb + (long long)hh * sBh;
  long long coff = (long long)bb * sCb + (long long)hh * sCh;
  C += coff;
  if (res) res += coff;

  int n0 = blockIdx.x * TBN;
  int m0 = blockIdx.y * TBM;
  int tid    = threadIdx.x;
  int wave   = tid >> 5;
  int lane   = tid & 31;
  int lanelo = lane & 15;
  int khalf  = lane >> 4;

  v8f acc[4] = {};

  const bool fast = (m0 + TBM <= M) && (n0 + TBN <= N) && ((K & (TBK - 1)) == 0);

  if (fast) {
    f4 ra[4], rb[4];

    auto loadA = [&](int kk) {
      #pragma unroll
      for (int s = 0; s < 4; ++s) {
        int idx = tid + (s << 7);
        int r = idx >> 3, c4 = (idx & 7) << 2;          // 8 float4 per 32-wide row
        ra[s] = *(const f4*)(A + (long long)(m0 + r) * lda + kk + c4);
      }
    };
    auto loadB = [&](int kk) {
      if (!transB) {
        #pragma unroll
        for (int s = 0; s < 4; ++s) {
          int idx = tid + (s << 7);
          int r = idx >> 4, c4 = (idx & 15) << 2;       // row = k, 16 float4 per 64-wide row
          rb[s] = *(const f4*)(Bm + (long long)(kk + r) * ldb + n0 + c4);
        }
      } else {
        #pragma unroll
        for (int s = 0; s < 4; ++s) {
          int idx = tid + (s << 7);
          int c = idx >> 3, r4 = (idx & 7) << 2;        // row = n, vector along K
          rb[s] = *(const f4*)(Bm + (long long)(n0 + c) * ldb + kk + r4);
        }
      }
    };
    auto stageLDS = [&]() {
      #pragma unroll
      for (int s = 0; s < 4; ++s) {
        int idx = tid + (s << 7);
        int r = idx >> 3, c4 = (idx & 7) << 2;
        h4 hv;
        hv[0] = (_Float16)ra[s][0]; hv[1] = (_Float16)ra[s][1];
        hv[2] = (_Float16)ra[s][2]; hv[3] = (_Float16)ra[s][3];
        *(h4*)&As[r][c4] = hv;                           // b64 store, 8B aligned
      }
      if (!transB) {
        #pragma unroll
        for (int s = 0; s < 4; ++s) {
          int idx = tid + (s << 7);
          int r = idx >> 4, c4 = (idx & 15) << 2;
          #pragma unroll
          for (int i = 0; i < 4; ++i)
            Bt[c4 + i][r] = (_Float16)rb[s][i];          // transpose into K-minor
        }
      } else {
        #pragma unroll
        for (int s = 0; s < 4; ++s) {
          int idx = tid + (s << 7);
          int c = idx >> 3, r4 = (idx & 7) << 2;
          h4 hv;
          hv[0] = (_Float16)rb[s][0]; hv[1] = (_Float16)rb[s][1];
          hv[2] = (_Float16)rb[s][2]; hv[3] = (_Float16)rb[s][3];
          *(h4*)&Bt[c][r4] = hv;                         // b64 store
        }
      }
    };

    loadA(0); loadB(0);
    for (int k0 = 0; k0 < K; k0 += TBK) {
      stageLDS();
      __syncthreads();
      if (k0 + TBK < K) { loadA(k0 + TBK); loadB(k0 + TBK); }  // overlap with WMMA

      // A fragment: pair (b32) reads, ISA 16-bit A 16x32 layout
      v16h af;
      const _Float16* arow = &As[wave * 16 + lanelo][0];
      #pragma unroll
      for (int j = 0; j < 8; ++j) {
        int kb = (j >> 2) * 16 + khalf * 8 + (j & 3) * 2;
        h2 t = *(const h2*)(arow + kb);
        af[2 * j] = t[0]; af[2 * j + 1] = t[1];
      }
      #pragma unroll
      for (int c = 0; c < 4; ++c) {
        v16h bf;
        const _Float16* brow = &Bt[c * 16 + lanelo][0];
        #pragma unroll
        for (int j = 0; j < 8; ++j) {
          int kb = (j >> 2) * 16 + khalf * 8 + (j & 3) * 2;
          h2 t = *(const h2*)(brow + kb);
          bf[2 * j] = t[0]; bf[2 * j + 1] = t[1];
        }
        acc[c] = __builtin_amdgcn_wmma_f32_16x16x32_f16(
            false, af, false, bf, (short)0, acc[c], false, false);
      }
      __syncthreads();
    }

    // unguarded epilogue
    #pragma unroll
    for (int c = 0; c < 4; ++c) {
      int col = n0 + c * 16 + lanelo;
      #pragma unroll
      for (int j = 0; j < 8; ++j) {
        int row = m0 + wave * 16 + khalf * 8 + j;
        float v = alpha * acc[c][j];
        if (bias)    v += bias[col];
        if (actGelu) v = gelu_tanh(v);
        if (res)     v += res[(long long)row * ldc + col];
        C[(long long)row * ldc + col] = v;
      }
    }
    return;
  }

  // ---------------- guarded slow path (partial tiles only) ----------------
  for (int k0 = 0; k0 < K; k0 += TBK) {
    #pragma unroll
    for (int s = 0; s < 16; ++s) {
      int idx = tid + 128 * s;
      int r = idx >> 5, c = idx & 31;
      int gm = m0 + r, gk = k0 + c;
      float v = (gm < M && gk < K) ? A[(long long)gm * lda + gk] : 0.0f;
      As[r][c] = (_Float16)v;
    }
    if (!transB) {
      #pragma unroll
      for (int s = 0; s < 16; ++s) {
        int idx = tid + 128 * s;
        int r = idx >> 6, c = idx & 63;
        int gk = k0 + r, gn = n0 + c;
        float v = (gk < K && gn < N) ? Bm[(long long)gk * ldb + gn] : 0.0f;
        Bt[c][r] = (_Float16)v;
      }
    } else {
      #pragma unroll
      for (int s = 0; s < 16; ++s) {
        int idx = tid + 128 * s;
        int c = idx >> 5, r = idx & 31;
        int gk = k0 + r, gn = n0 + c;
        float v = (gk < K && gn < N) ? Bm[(long long)gn * ldb + gk] : 0.0f;
        Bt[c][r] = (_Float16)v;
      }
    }
    __syncthreads();

    v16h af;
    const _Float16* arow = &As[wave * 16 + lanelo][0];
    #pragma unroll
    for (int j = 0; j < 8; ++j) {
      int kb = (j >> 2) * 16 + khalf * 8 + (j & 3) * 2;
      h2 t = *(const h2*)(arow + kb);
      af[2 * j] = t[0]; af[2 * j + 1] = t[1];
    }
    #pragma unroll
    for (int c = 0; c < 4; ++c) {
      v16h bf;
      const _Float16* brow = &Bt[c * 16 + lanelo][0];
      #pragma unroll
      for (int j = 0; j < 8; ++j) {
        int kb = (j >> 2) * 16 + khalf * 8 + (j & 3) * 2;
        h2 t = *(const h2*)(brow + kb);
        bf[2 * j] = t[0]; bf[2 * j + 1] = t[1];
      }
      acc[c] = __builtin_amdgcn_wmma_f32_16x16x32_f16(
          false, af, false, bf, (short)0, acc[c], false, false);
    }
    __syncthreads();
  }

  #pragma unroll
  for (int c = 0; c < 4; ++c) {
    int col = n0 + c * 16 + lanelo;
    #pragma unroll
    for (int j = 0; j < 8; ++j) {
      int row = m0 + wave * 16 + khalf * 8 + j;
      if (row < M && col < N) {
        float v = alpha * acc[c][j];
        if (bias)    v += bias[col];
        if (actGelu) v = gelu_tanh(v);
        if (res)     v += res[(long long)row * ldc + col];
        C[(long long)row * ldc + col] = v;
      }
    }
  }
}

// ---------------------------------------------------------------------------
__global__ void k_embed(const int* __restrict__ tok, const float* __restrict__ emb,
                        float* __restrict__ x, int Nn, int D) {
  int idx = blockIdx.x * blockDim.x + threadIdx.x;
  if (idx >= Nn * D) return;
  int n = idx / D, d = idx - n * D;
  x[idx] = emb[(long long)tok[n] * D + d];
}

__global__ void k_zero(float* __restrict__ p, int n) {
  int idx = blockIdx.x * blockDim.x + threadIdx.x;
  if (idx < n) p[idx] = 0.0f;
}

__global__ void k_layernorm(const float* __restrict__ x, const float* __restrict__ w,
                            const float* __restrict__ b, float* __restrict__ y, int D) {
  int row = blockIdx.x;
  const float* xr = x + (long long)row * D;
  float* yr = y + (long long)row * D;
  __shared__ float s1[256], s2[256];
  int t = threadIdx.x;
  float a = 0.f, aq = 0.f;
  for (int i = t; i < D; i += 256) { float v = xr[i]; a += v; aq += v * v; }
  s1[t] = a; s2[t] = aq;
  __syncthreads();
  for (int off = 128; off > 0; off >>= 1) {
    if (t < off) { s1[t] += s1[t + off]; s2[t] += s2[t + off]; }
    __syncthreads();
  }
  float mu = s1[0] / D;
  float var = s2[0] / D - mu * mu;
  float rstd = rsqrtf(var + 1e-5f);
  for (int i = t; i < D; i += 256)
    yr[i] = (xr[i] - mu) * rstd * w[i] + b[i];
}

__global__ void k_pos_trunk(float* __restrict__ pos, int Nn, int T) {
  int n = blockIdx.x * blockDim.x + threadIdx.x;
  if (n < Nn) pos[n] = (float)(n % T);
}

__global__ void k_epos(const int* __restrict__ ids, int e, float* __restrict__ pos, int Nn) {
  if (threadIdx.x == 0 && blockIdx.x == 0) {
    int cnt = 0;
    for (int n = 0; n < Nn; ++n) {
      cnt += (ids[n] == e) ? 1 : 0;
      int p = cnt - 1;
      pos[n] = (float)(p > 0 ? p : 0);
    }
  }
}

// in-place RoPE on q,k laid out [N, H*hd]; one thread per (n,h,i<hd/2)
__global__ void k_rope(float* __restrict__ q, float* __restrict__ k,
                       const float* __restrict__ pos, int Nn, int H, int hd) {
  int half = hd >> 1;
  int idx = blockIdx.x * blockDim.x + threadIdx.x;
  int total = Nn * H * half;
  if (idx >= total) return;
  int n = idx / (H * half);
  int rem = idx - n * (H * half);
  int h = rem / half, i = rem - h * half;
  long long base = (long long)n * (H * hd) + (long long)h * hd;
  float p = pos[n];
  float inv = powf(10000.0f, -((float)(2 * i) / (float)hd));
  float f = p * inv;
  float c = cosf(f), s = sinf(f);
  float q1 = q[base + i], q2 = q[base + i + half];
  q[base + i]        = q1 * c - q2 * s;
  q[base + i + half] = q2 * c + q1 * s;
  float k1 = k[base + i], k2 = k[base + i + half];
  k[base + i]        = k1 * c - k2 * s;
  k[base + i + half] = k2 * c + k1 * s;
}

// masked softmax over rows of scores [Z, Tq, Tk]; grid=(Tq, Z), block=256
// mode 0: causal (col<=row). mode 1: expert mask via ids/e with diagonal kept.
__global__ void k_softmax(float* __restrict__ sc, int Tk, int mode,
                          const int* __restrict__ ids, int e) {
  int row = blockIdx.x;
  int z = blockIdx.y;
  int Tq = gridDim.x;
  float* r = sc + ((long long)z * Tq + row) * Tk;
  __shared__ float red[256];
  int t = threadIdx.x;
  int idrow = (mode == 1) ? ids[row] : 0;

  float mx = -3.4e38f;
  for (int j = t; j < Tk; j += 256) {
    bool allowed = (mode == 0)
        ? (j <= row)
        : (((idrow == e) && (ids[j] == e) && (j <= row)) || (j == row));
    float v = r[j] + (allowed ? 0.0f : -1e9f);
    mx = fmaxf(mx, v);
  }
  red[t] = mx; __syncthreads();
  for (int off = 128; off > 0; off >>= 1) {
    if (t < off) red[t] = fmaxf(red[t], red[t + off]);
    __syncthreads();
  }
  mx = red[0];
  __syncthreads();

  float sum = 0.f;
  for (int j = t; j < Tk; j += 256) {
    bool allowed = (mode == 0)
        ? (j <= row)
        : (((idrow == e) && (ids[j] == e) && (j <= row)) || (j == row));
    float v = r[j] + (allowed ? 0.0f : -1e9f);
    float p = expf(v - mx);
    r[j] = p;
    sum += p;
  }
  red[t] = sum; __syncthreads();
  for (int off = 128; off > 0; off >>= 1) {
    if (t < off) red[t] += red[t + off];
    __syncthreads();
  }
  float invs = 1.0f / red[0];
  for (int j = t; j < Tk; j += 256) r[j] *= invs;
}

__global__ void k_router(const float* __restrict__ rl, float* __restrict__ probs,
                         float* __restrict__ idsf, int* __restrict__ ids, int Nn, int E) {
  int n = blockIdx.x * blockDim.x + threadIdx.x;
  if (n >= Nn) return;
  const float* r = rl + (long long)n * E;
  float mx = r[0]; int am = 0;
  for (int j = 1; j < E; ++j) if (r[j] > mx) { mx = r[j]; am = j; }
  float s = 0.f;
  for (int j = 0; j < E; ++j) s += expf(r[j] - mx);
  float invs = 1.0f / s;
  for (int j = 0; j < E; ++j) probs[(long long)n * E + j] = expf(r[j] - mx) * invs;
  ids[n] = am;
  idsf[n] = (float)am;
}

__global__ void k_scatter(const float* __restrict__ src, float* __restrict__ dst,
                          const int* __restrict__ ids, int e, int Nn, int D) {
  int idx = blockIdx.x * blockDim.x + threadIdx.x;
  if (idx >= Nn * D) return;
  int n = idx / D;
  if (ids[n] == e) dst[idx] = src[idx];
}

// ---------------------------------------------------------------------------
extern "C" void kernel_launch(void* const* d_in, const int* in_sizes, int n_in,
                              void* d_out, int out_size, void* d_ws, size_t ws_size,
                              hipStream_t stream) {
  const int Bn = 2, T = 512, Nn = 1024, Dd = 1024, Hh = 16, hd = 64;
  const int LT = 4, LE = 4, Ee = 9, FT = 4096, FE = 2048, RH = 256, Vv = 32000;

  const int*   tokens  = (const int*)d_in[0];
  const float* emb     = (const float*)d_in[1];
  const float* t_ln1_w = (const float*)d_in[2];
  const float* t_ln1_b = (const float*)d_in[3];
  const float* t_wq    = (const float*)d_in[4];
  const float* t_wk    = (const float*)d_in[5];
  const float* t_wv    = (const float*)d_in[6];
  const float* t_wo    = (const float*)d_in[7];
  const float* t_ln2_w = (const float*)d_in[8];
  const float* t_ln2_b = (const float*)d_in[9];
  const float* t_fc1   = (const float*)d_in[10];
  const float* t_fc2   = (const float*)d_in[11];
  const float* t_nw    = (const float*)d_in[12];
  const float* t_nb    = (const float*)d_in[13];
  const float* r_w1    = (const float*)d_in[14];
  const float* r_b1    = (const float*)d_in[15];
  const float* r_w2    = (const float*)d_in[16];
  const float* r_b2    = (const float*)d_in[17];
  const float* e_ln1_w = (const float*)d_in[18];
  const float* e_ln1_b = (const float*)d_in[19];
  const float* e_wq    = (const float*)d_in[20];
  const float* e_wk    = (const float*)d_in[21];
  const float* e_wv    = (const float*)d_in[22];
  const float* e_wo    = (const float*)d_in[23];
  const float* e_ln2_w = (const float*)d_in[24];
  const float* e_ln2_b = (const float*)d_in[25];
  const float* e_fc1   = (const float*)d_in[26];
  const float* e_fc2   = (const float*)d_in[27];
  const float* e_nw    = (const float*)d_in[28];
  const float* e_nb    = (const float*)d_in[29];
  (void)in_sizes; (void)n_in; (void)out_size; (void)ws_size;

  float* out_f = (float*)d_out;
  float* logits = out_f;                                    // [N, V]
  float* probs  = out_f + (long long)Nn * Vv;               // [N, E]
  float* rlog   = probs + (long long)Nn * Ee;               // [N, E]
  float* idsf   = rlog + (long long)Nn * Ee;                // [N]

  // workspace bump allocator
  char* wsp = (char*)d_ws;
  auto alloc = [&](size_t bytes) -> void* {
    void* p = (void*)wsp;
    wsp += (bytes + 255) & ~(size_t)255;
    return p;
  };
  const size_t ND = (size_t)Nn * Dd;
  float* x      = (float*)alloc(ND * 4);
  float* xn     = (float*)alloc(ND * 4);
  float* qb     = (float*)alloc(ND * 4);
  float* kb     = (float*)alloc(ND * 4);
  float* vb     = (float*)alloc(ND * 4);
  float* yb     = (float*)alloc(ND * 4);
  float* hf     = (float*)alloc(ND * 4);
  float* xe     = (float*)alloc(ND * 4);
  float* outbuf = (float*)alloc(ND * 4);
  float* scores = (float*)alloc((size_t)Hh * Nn * Nn * 4);  // 64 MB, covers trunk too
  float* ffn    = (float*)alloc((size_t)Nn * FT * 4);
  float* pos    = (float*)alloc((size_t)Nn * 4);
  int*   idsi   = (int*)alloc((size_t)Nn * 4);

  auto gemm = [&](const float* A, const float* Bm, float* C,
                  const float* bias, const float* res,
                  int M, int N, int K, int lda, int ldb, int ldc,
                  int transB, int act, float alpha,
                  int batch, int Hdim,
                  long long sAb, long long sAh, long long sBb, long long sBh,
                  long long sCb, long long sCh) {
    dim3 g((N + TBN - 1) / TBN, (M + TBM - 1) / TBM, batch);
    k_gemm<<<g, 128, 0, stream>>>(A, Bm, C, bias, res, M, N, K, lda, ldb, ldc,
                                  sAb, sAh, sBb, sBh, sCb, sCh,
                                  Hdim, transB, act, alpha);
  };

  const float scale = 0.125f;  // 1/sqrt(64)

  // one full transformer layer operating on xbuf [N,D]
  auto tlayer = [&](float* xbuf,
                    const float* ln1w, const float* ln1b,
                    const float* wq, const float* wk, const float* wv, const float* wo,
                    const float* ln2w, const float* ln2b,
                    const float* fc1, const float* fc2, int Fdim,
                    int nBatch, int seqT, int smode, const int* smids, int se) {
    // xn = LN(x)
    k_layernorm<<<Nn, 256, 0, stream>>>(xbuf, ln1w, ln1b, xn, Dd);
    // q,k,v
    gemm(xn, wq, qb, nullptr, nullptr, Nn, Dd, Dd, Dd, Dd, Dd, 0, 0, 1.0f, 1, 1, 0,0,0,0,0,0);
    gemm(xn, wk, kb, nullptr, nullptr, Nn, Dd, Dd, Dd, Dd, Dd, 0, 0, 1.0f, 1, 1, 0,0,0,0,0,0);
    gemm(xn, wv, vb, nullptr, nullptr, Nn, Dd, Dd, Dd, Dd, Dd, 0, 0, 1.0f, 1, 1, 0,0,0,0,0,0);
    // rope
    {
      int total = Nn * Hh * (hd / 2);
      k_rope<<<(total + 255) / 256, 256, 0, stream>>>(qb, kb, pos, Nn, Hh, hd);
    }
    int BH = nBatch * Hh;
    long long sqT = (long long)seqT * Dd;                 // per-batch stride in q/k/v
    long long sS  = (long long)seqT * seqT;               // per-head stride in scores
    // scores = (Q @ K^T) * scale
    gemm(qb, kb, scores, nullptr, nullptr, seqT, seqT, hd, Dd, Dd, seqT,
         1, 0, scale, BH, Hh, sqT, hd, sqT, hd, (long long)Hh * sS, sS);
    // masked softmax in place
    k_softmax<<<dim3(seqT, BH), 256, 0, stream>>>(scores, seqT, smode, smids, se);
    // y = P @ V  (written back into head-interleaved [N, D])
    gemm(scores, vb, yb, nullptr, nullptr, seqT, hd, seqT, seqT, Dd, Dd,
         0, 0, 1.0f, BH, Hh, (long long)Hh * sS, sS, sqT, hd, sqT, hd);
    // x = y @ wo + x
    gemm(yb, wo, xbuf, nullptr, xbuf, Nn, Dd, Dd, Dd, Dd, Dd, 0, 0, 1.0f, 1, 1, 0,0,0,0,0,0);
    // FFN
    k_layernorm<<<Nn, 256, 0, stream>>>(xbuf, ln2w, ln2b, xn, Dd);
    gemm(xn, fc1, ffn, nullptr, nullptr, Nn, Fdim, Dd, Dd, Fdim, Fdim,
         0, 1 /*gelu*/, 1.0f, 1, 1, 0,0,0,0,0,0);
    gemm(ffn, fc2, xbuf, nullptr, xbuf, Nn, Dd, Fdim, Fdim, Dd, Dd,
         0, 0, 1.0f, 1, 1, 0,0,0,0,0,0);
  };

  // ---- trunk ----
  k_embed<<<(Nn * Dd + 255) / 256, 256, 0, stream>>>(tokens, emb, x, Nn, Dd);
  k_pos_trunk<<<(Nn + 255) / 256, 256, 0, stream>>>(pos, Nn, T);
  for (int l = 0; l < LT; ++l) {
    tlayer(x,
           t_ln1_w + (size_t)l * Dd, t_ln1_b + (size_t)l * Dd,
           t_wq + (size_t)l * Dd * Dd, t_wk + (size_t)l * Dd * Dd,
           t_wv + (size_t)l * Dd * Dd, t_wo + (size_t)l * Dd * Dd,
           t_ln2_w + (size_t)l * Dd, t_ln2_b + (size_t)l * Dd,
           t_fc1 + (size_t)l * Dd * FT, t_fc2 + (size_t)l * FT * Dd, FT,
           Bn, T, 0, nullptr, 0);
  }
  k_layernorm<<<Nn, 256, 0, stream>>>(x, t_nw, t_nb, hf, Dd);

  // ---- router ----
  gemm(hf, r_w1, ffn, r_b1, nullptr, Nn, RH, Dd, Dd, RH, RH, 0, 1, 1.0f, 1, 1, 0,0,0,0,0,0);
  gemm(ffn, r_w2, rlog, r_b2, nullptr, Nn, Ee, RH, RH, Ee, Ee, 0, 0, 1.0f, 1, 1, 0,0,0,0,0,0);
  k_router<<<(Nn + 255) / 256, 256, 0, stream>>>(rlog, probs, idsf, idsi, Nn, Ee);

  // ---- experts ----
  k_zero<<<(Nn * Dd + 255) / 256, 256, 0, stream>>>(outbuf, Nn * Dd);
  for (int e = 0; e < Ee; ++e) {
    k_epos<<<1, 32, 0, stream>>>(idsi, e, pos, Nn);
    hipMemcpyAsync(xe, hf, ND * sizeof(float), hipMemcpyDeviceToDevice, stream);
    for (int l = 0; l < LE; ++l) {
      size_t el = (size_t)e * LE + l;
      tlayer(xe,
             e_ln1_w + el * Dd, e_ln1_b + el * Dd,
             e_wq + el * Dd * Dd, e_wk + el * Dd * Dd,
             e_wv + el * Dd * Dd, e_wo + el * Dd * Dd,
             e_ln2_w + el * Dd, e_ln2_b + el * Dd,
             e_fc1 + el * Dd * FE, e_fc2 + el * FE * Dd, FE,
             1, Nn, 1, idsi, e);
    }
    k_layernorm<<<Nn, 256, 0, stream>>>(xe, e_nw + (size_t)e * Dd, e_nb + (size_t)e * Dd, xn, Dd);
    k_scatter<<<(Nn * Dd + 255) / 256, 256, 0, stream>>>(xn, outbuf, idsi, e, Nn, Dd);
  }

  // ---- tied lm head: logits = out @ emb^T ----
  gemm(outbuf, emb, logits, nullptr, nullptr, Nn, Vv, Dd, Dd, Dd, Vv,
       1 /*transB*/, 0, 1.0f, 1, 1, 0,0,0,0,0,0);
}
